// MultiModalConsensusMoE_19250043421312
// MI455X (gfx1250) — matrix operand order repaired and checked by reading, not compiled, not run on hardware
//
#include <hip/hip_runtime.h>
#include <math.h>

typedef __bf16 bf16_t;
typedef __attribute__((ext_vector_type(16))) __bf16 v16bf;
typedef __attribute__((ext_vector_type(8)))  float  v8f;

__device__ __forceinline__ float gelu_erf(float x) {
    return 0.5f * x * (1.0f + erff(x * 0.70710678118654752f));
}

// LDS aperture keeps the workgroup-relative byte offset in addr[31:0] (ISA 10.2),
// so truncating the flat address of a __shared__ object yields the LDS offset the
// async-to-LDS instructions expect in VDST.
__device__ __forceinline__ unsigned lds_off(const void* p) {
    return (unsigned)(unsigned long long)p;
}

// One CDNA5 async global->LDS 16B transfer per lane (tracked by ASYNCcnt).
__device__ __forceinline__ void async_b128(unsigned loff, const bf16_t* g) {
    asm volatile("global_load_async_to_lds_b128 %0, %1, off"
                 :: "v"(loff), "v"(g) : "memory");
}
__device__ __forceinline__ void wait_async0() {
    asm volatile("s_wait_asynccnt 0" ::: "memory");
}

// ---------------- fp32 -> bf16 conversion (row-major, for A-side activations) ----------------
__global__ void cvt_f32_bf16(const float* __restrict__ src, bf16_t* __restrict__ dst, long n) {
    long i = (long)blockIdx.x * blockDim.x + threadIdx.x;
    long stride = (long)gridDim.x * blockDim.x;
    for (; i < n; i += stride) dst[i] = (bf16_t)src[i];
}

// ---------------- fp32 [K,N] -> bf16 [N,K] transpose-convert (for all B-side weights) ----------------
__global__ void cvt_T_f32_bf16(const float* __restrict__ src, bf16_t* __restrict__ dst,
                               int K, int N) {
    int total = K * N;
    for (int i = blockIdx.x * blockDim.x + threadIdx.x; i < total;
         i += gridDim.x * blockDim.x) {
        const int k = i / N, n = i % N;
        dst[(size_t)n * K + k] = (bf16_t)src[i];
    }
}

// ---------------- WMMA GEMM ----------------
// C = A[M,K](bf16,row-major) * Bt[N,K](bf16, B pre-transposed) (+bias[N]).
// Block tile 128x128, BK=32 (one v_wmma_f32_16x16x32_bf16 K-step).
// Double-buffered LDS staged entirely with global_load_async_to_lds_b128.
// 4 waves (128 thr); each wave owns a 64x64 quadrant = 4x4 WMMA tiles
// -> 16 WMMAs per 16 ds_load_b128 per K-step (1.0 LDS-load per WMMA).
enum { EPI_BIAS_BF16 = 0, EPI_GELU_BF16 = 1, EPI_SCALE_ACC_F32 = 2 };

template<int EPI>
__global__ __launch_bounds__(128) void gemm_bf16(
    const bf16_t* __restrict__ A, const bf16_t* __restrict__ Bt,
    const float* __restrict__ bias, void* __restrict__ Dv,
    const float* __restrict__ scale, int M, int N, int K,
    int sstride, int accumulate)
{
    // row stride 48 bf16 = 96 B: 16B-aligned rows, staggered bank phase
    __shared__ alignas(16) bf16_t As[2][128][48];   // As[buf][m][k]
    __shared__ alignas(16) bf16_t Bs[2][128][48];   // Bs[buf][n][k]

    const int t    = threadIdx.x;
    const int lane = t & 31;
    const int wv   = t >> 5;        // 0..3
    const int wm   = wv >> 1;       // 0..1 -> 64 rows each
    const int wn   = wv & 1;        // 0..1 -> 64 cols each
    const int bm   = blockIdx.y * 128;
    const int bn   = blockIdx.x * 128;

    v8f acc[4][4];
    const v8f vzero = {0.f,0.f,0.f,0.f,0.f,0.f,0.f,0.f};
#pragma unroll
    for (int i = 0; i < 4; ++i)
#pragma unroll
        for (int j = 0; j < 4; ++j) acc[i][j] = vzero;

    // each thread owns one full 32-element K-row of the A tile and of the Bt tile
    const bf16_t* gA = A  + (size_t)(bm + t) * K;
    const bf16_t* gB = Bt + (size_t)(bn + t) * K;

    auto issue = [&](int buf, int k0) {
#pragma unroll
        for (int s = 0; s < 4; ++s) {
            async_b128(lds_off(&As[buf][t][s * 8]), gA + k0 + s * 8);
            async_b128(lds_off(&Bs[buf][t][s * 8]), gB + k0 + s * 8);
        }
    };

    issue(0, 0);
    int buf = 0;
    for (int k0 = 0; k0 < K; k0 += 32) {
        wait_async0();
        __syncthreads();
        if (k0 + 32 < K) issue(buf ^ 1, k0 + 32);

        // --- fragment fill (contiguous k = 16h + s, same order in A and B), packed u32 (2xbf16) LDS loads ---
        const int kb  = (lane >> 4) * 8;    // unused; A reads use kb2 like B
        const int kb2 = (lane >> 4) * 16;   // B: lanes 16-31 start at K=16
        v16bf afrag[4];
#pragma unroll
        for (int tm = 0; tm < 4; ++tm) {
            unsigned int* ap = (unsigned int*)&afrag[tm];
            const int ar = wm * 64 + tm * 16 + (lane & 15);
#pragma unroll
            for (int j = 0; j < 8; ++j)
                ap[j] = *(const unsigned int*)&As[buf][ar][kb2 + 2 * j];
        }
        v16bf bfrag[4];
#pragma unroll
        for (int tn = 0; tn < 4; ++tn) {
            unsigned int* bp = (unsigned int*)&bfrag[tn];
            const int bc = wn * 64 + tn * 16 + (lane & 15);
#pragma unroll
            for (int j = 0; j < 8; ++j)
                bp[j] = *(const unsigned int*)&Bs[buf][bc][kb2 + 2 * j];
        }
#pragma unroll
        for (int tm = 0; tm < 4; ++tm)
#pragma unroll
            for (int tn = 0; tn < 4; ++tn)
                acc[tm][tn] = __builtin_amdgcn_wmma_f32_16x16x32_bf16(
                    false, afrag[tm], false, bfrag[tn],
                    (short)0, acc[tm][tn], false, false);
        buf ^= 1;
    }

    // --- epilogue: C/D layout = lanes 0-15: M=r, lanes 16-31: M=8+r; N=lane&15 ---
    bf16_t* D16 = (bf16_t*)Dv;
    float*  D32 = (float*)Dv;
#pragma unroll
    for (int tm = 0; tm < 4; ++tm) {
        const int row0 = bm + wm * 64 + tm * 16 + (lane >> 4) * 8;
#pragma unroll
        for (int tn = 0; tn < 4; ++tn) {
            const int col = bn + wn * 64 + tn * 16 + (lane & 15);
            const float bc = bias ? bias[col] : 0.f;
#pragma unroll
            for (int r = 0; r < 8; ++r) {
                const int row = row0 + r;
                float v = acc[tm][tn][r] + bc;
                if (EPI == EPI_BIAS_BF16) {
                    D16[(size_t)row * N + col] = (bf16_t)v;
                } else if (EPI == EPI_GELU_BF16) {
                    D16[(size_t)row * N + col] = (bf16_t)gelu_erf(v);
                } else {
                    const float s = scale[(size_t)row * sstride];
                    float* dp = &D32[(size_t)row * N + col];
                    *dp = accumulate ? (*dp + s * v) : (s * v);
                }
            }
        }
    }
}

// ---------------- router: logits = act[M,512](bf16) @ rw2[512,8] + rb2, softmax over 8 ----------------
__global__ __launch_bounds__(256) void route_softmax(
    const bf16_t* __restrict__ act, const float* __restrict__ rw2,
    const float* __restrict__ rb2, float* __restrict__ route_w, int M)
{
    const int row = blockIdx.x * blockDim.x + threadIdx.x;
    if (row >= M) return;
    float acc[8];
#pragma unroll
    for (int e = 0; e < 8; ++e) acc[e] = rb2[e];
    const bf16_t* a = act + (size_t)row * 512;
    for (int k = 0; k < 512; ++k) {
        const float av = (float)a[k];
#pragma unroll
        for (int e = 0; e < 8; ++e) acc[e] += av * rw2[k * 8 + e];
    }
    float mx = acc[0];
#pragma unroll
    for (int e = 1; e < 8; ++e) mx = fmaxf(mx, acc[e]);
    float s = 0.f;
#pragma unroll
    for (int e = 0; e < 8; ++e) { acc[e] = expf(acc[e] - mx); s += acc[e]; }
    const float inv = 1.f / s;
#pragma unroll
    for (int e = 0; e < 8; ++e) route_w[(size_t)row * 8 + e] = acc[e] * inv;
}

// ---------------- modality score: mlogits[row,m] = act[M,512](bf16) @ sw2[512,1] + sb2 ----------------
__global__ __launch_bounds__(256) void score_logit(
    const bf16_t* __restrict__ act, const float* __restrict__ sw2,
    const float* __restrict__ sb2, float* __restrict__ mlogits, int M, int m)
{
    const int row = blockIdx.x * blockDim.x + threadIdx.x;
    if (row >= M) return;
    const bf16_t* a = act + (size_t)row * 512;
    float acc = sb2[0];
    for (int k = 0; k < 512; ++k) acc += (float)a[k] * sw2[k];
    mlogits[(size_t)row * 4 + m] = acc;
}

// ---------------- top-2 over 4 modalities + softmax + consensus + LayerNorm ----------------
__global__ __launch_bounds__(256) void consensus_ln(
    const float* __restrict__ mixed,    // [4, M, 512]
    const float* __restrict__ mlogits,  // [M, 4]
    const float* __restrict__ ln_g, const float* __restrict__ ln_b,
    float* __restrict__ out, int M)
{
    const int lane = threadIdx.x & 31;
    const int wv   = threadIdx.x >> 5;
    const int row  = blockIdx.x * 8 + wv;
    if (row >= M) return;

    float l[4];
#pragma unroll
    for (int i = 0; i < 4; ++i) l[i] = mlogits[(size_t)row * 4 + i];
    int i1 = 0;
#pragma unroll
    for (int i = 1; i < 4; ++i) if (l[i] > l[i1]) i1 = i;   // earliest max (jax tie-break)
    int i2 = -1;
#pragma unroll
    for (int i = 0; i < 4; ++i) {
        if (i == i1) continue;
        if (i2 < 0 || l[i] > l[i2]) i2 = i;
    }
    float w[4]; float ssum = 0.f;
#pragma unroll
    for (int i = 0; i < 4; ++i) {
        w[i] = (i == i1 || i == i2) ? expf(l[i] - l[i1]) : 0.f;
        ssum += w[i];
    }
    const float winv = 1.f / ssum;

    float c[16];
#pragma unroll
    for (int j = 0; j < 16; ++j) {
        const int col = j * 32 + lane;
        float v = 0.f;
#pragma unroll
        for (int m = 0; m < 4; ++m)
            v += w[m] * mixed[((size_t)m * M + row) * 512 + col];
        c[j] = v * winv;
    }
    float sum = 0.f, sq = 0.f;
#pragma unroll
    for (int j = 0; j < 16; ++j) { sum += c[j]; sq += c[j] * c[j]; }
#pragma unroll
    for (int off = 16; off > 0; off >>= 1) {
        sum += __shfl_xor(sum, off, 32);
        sq  += __shfl_xor(sq,  off, 32);
    }
    const float mu   = sum * (1.f / 512.f);
    const float var  = sq * (1.f / 512.f) - mu * mu;
    const float rinv = rsqrtf(var + 1e-5f);
#pragma unroll
    for (int j = 0; j < 16; ++j) {
        const int col = j * 32 + lane;
        out[(size_t)row * 512 + col] = (c[j] - mu) * rinv * ln_g[col] + ln_b[col];
    }
}

// ---------------- orchestration ----------------
extern "C" void kernel_launch(void* const* d_in, const int* in_sizes, int n_in,
                              void* d_out, int out_size, void* d_ws, size_t ws_size,
                              hipStream_t stream)
{
    (void)in_sizes; (void)n_in; (void)out_size; (void)ws_size;
    const int M = 4096, H = 512, E = 8;
    const int dims[4] = {1024, 64, 768, 512};

    const float* feat[4]; const float* pw[4]; const float* pb[4];
    for (int m = 0; m < 4; ++m) {
        feat[m] = (const float*)d_in[3 * m + 0];
        pw[m]   = (const float*)d_in[3 * m + 1];
        pb[m]   = (const float*)d_in[3 * m + 2];
    }
    const float* ew1 = (const float*)d_in[12];
    const float* eb1 = (const float*)d_in[13];
    const float* ew2 = (const float*)d_in[14];
    const float* eb2 = (const float*)d_in[15];
    const float* rw1 = (const float*)d_in[16];
    const float* rb1 = (const float*)d_in[17];
    const float* rw2 = (const float*)d_in[18];
    const float* rb2 = (const float*)d_in[19];
    const float* sw1 = (const float*)d_in[20];
    const float* sb1 = (const float*)d_in[21];
    const float* sw2 = (const float*)d_in[22];
    const float* sb2 = (const float*)d_in[23];
    const float* lng = (const float*)d_in[24];
    const float* lnb = (const float*)d_in[25];

    // workspace carve-out (everything stays L2-resident: ~81 MB << 192 MB)
    char* p = (char*)d_ws;
    auto alloc = [&](size_t bytes) -> void* {
        void* r = (void*)p;
        p += (bytes + 255) & ~(size_t)255;
        return r;
    };
    bf16_t* feat16[4]; bf16_t* pwT16[4];
    for (int m = 0; m < 4; ++m) {
        feat16[m] = (bf16_t*)alloc((size_t)M * dims[m] * sizeof(bf16_t));
        pwT16[m]  = (bf16_t*)alloc((size_t)dims[m] * H * sizeof(bf16_t));
    }
    bf16_t* rw1T16 = (bf16_t*)alloc((size_t)H * H * sizeof(bf16_t));
    bf16_t* sw1T16 = (bf16_t*)alloc((size_t)H * H * sizeof(bf16_t));
    bf16_t* ew1T16 = (bf16_t*)alloc((size_t)E * H * H * sizeof(bf16_t));
    bf16_t* ew2T16 = (bf16_t*)alloc((size_t)E * H * H * sizeof(bf16_t));
    bf16_t* X16    = (bf16_t*)alloc((size_t)M * H * sizeof(bf16_t));
    bf16_t* H1_16  = (bf16_t*)alloc((size_t)M * H * sizeof(bf16_t));
    bf16_t* ACT16  = (bf16_t*)alloc((size_t)M * H * sizeof(bf16_t));
    bf16_t* MIX16  = (bf16_t*)alloc((size_t)M * H * sizeof(bf16_t));
    float*  mixed  = (float*)alloc((size_t)4 * M * H * sizeof(float));
    float*  routew = (float*)alloc((size_t)M * E * sizeof(float));
    float*  mlog   = (float*)alloc((size_t)M * 4 * sizeof(float));

    auto cvt = [&](const float* s, bf16_t* d, long n) {
        cvt_f32_bf16<<<512, 256, 0, stream>>>(s, d, n);
    };
    auto cvtT = [&](const float* s, bf16_t* d, int K, int N) {
        cvt_T_f32_bf16<<<256, 256, 0, stream>>>(s, d, K, N);
    };
    for (int m = 0; m < 4; ++m) {
        cvt(feat[m], feat16[m], (long)M * dims[m]);
        cvtT(pw[m], pwT16[m], dims[m], H);
    }
    cvtT(rw1, rw1T16, H, H);
    cvtT(sw1, sw1T16, H, H);
    for (int e = 0; e < E; ++e) {
        cvtT(ew1 + (size_t)e * H * H, ew1T16 + (size_t)e * H * H, H, H);
        cvtT(ew2 + (size_t)e * H * H, ew2T16 + (size_t)e * H * H, H, H);
    }

    const dim3 gblk(128);
    const dim3 ggrid(H / 128, M / 128);   // (4, 32)

    for (int m = 0; m < 4; ++m) {
        // X = feat @ pw + pb  -> bf16
        gemm_bf16<EPI_BIAS_BF16><<<ggrid, gblk, 0, stream>>>(
            feat16[m], pwT16[m], pb[m], X16, nullptr, M, H, dims[m], 0, 0);
        // router hidden: gelu(X @ rw1 + rb1) -> bf16
        gemm_bf16<EPI_GELU_BF16><<<ggrid, gblk, 0, stream>>>(
            X16, rw1T16, rb1, ACT16, nullptr, M, H, H, 0, 0);
        route_softmax<<<M / 256, 256, 0, stream>>>(ACT16, rw2, rb2, routew, M);

        float* mix_m = mixed + (size_t)m * M * H;
        for (int e = 0; e < 8; ++e) {
            // h_e = gelu(X @ ew1[e] + eb1[e]) -> bf16
            gemm_bf16<EPI_GELU_BF16><<<ggrid, gblk, 0, stream>>>(
                X16, ew1T16 + (size_t)e * H * H, eb1 + e * H, H1_16, nullptr, M, H, H, 0, 0);
            // mixed[m] (+)= route_w[:,e] * (h_e @ ew2[e] + eb2[e])  -> fp32
            gemm_bf16<EPI_SCALE_ACC_F32><<<ggrid, gblk, 0, stream>>>(
                H1_16, ew2T16 + (size_t)e * H * H, eb2 + e * H, mix_m,
                routew + e, M, H, H, E, e > 0 ? 1 : 0);
        }
        // score: gelu(mixed @ sw1 + sb1) @ sw2 + sb2
        cvt(mix_m, MIX16, (long)M * H);
        gemm_bf16<EPI_GELU_BF16><<<ggrid, gblk, 0, stream>>>(
            MIX16, sw1T16, sb1, ACT16, nullptr, M, H, H, 0, 0);
        score_logit<<<M / 256, 256, 0, stream>>>(ACT16, sw2, sb2, mlog, M, m);
    }

    consensus_ln<<<M / 8, 256, 0, stream>>>(mixed, mlog, lng, lnb, (float*)d_out, M);
}
